// TransformerBlock_89163521065465
// MI455X (gfx1250) — matrix operand order repaired and checked by reading, not compiled, not run on hardware
//
#include <hip/hip_runtime.h>

// ---------- types ----------
typedef __attribute__((ext_vector_type(16))) __bf16        v16bf;
typedef __attribute__((ext_vector_type(8)))  float         v8f;
typedef __attribute__((ext_vector_type(4)))  unsigned int  u32x4;
typedef __attribute__((ext_vector_type(4)))  float         f32x4;
typedef __attribute__((ext_vector_type(4)))  unsigned short u16x4;

union Frag32 { u32x4 u[2]; v16bf v; unsigned short s[16]; };

__device__ __forceinline__ unsigned short f2bf(float f) {
  unsigned int u = __float_as_uint(f);
  u += 0x7FFFu + ((u >> 16) & 1u);          // round-to-nearest-even
  return (unsigned short)(u >> 16);
}
__device__ __forceinline__ v8f vzero8() {
  v8f z = {0.f,0.f,0.f,0.f,0.f,0.f,0.f,0.f};
  return z;
}

// CDNA5 async global->LDS direct copy (bypasses VGPRs, tracked by ASYNCcnt).
__device__ __forceinline__ void async_ld_b128(unsigned lds_off, const void* g) {
  asm volatile("global_load_async_to_lds_b128 %0, %1, off"
               :: "v"(lds_off), "v"(g) : "memory");
}
#define WAIT_ASYNC(imm) asm volatile("s_wait_asynccnt " #imm ::: "memory")
__device__ __forceinline__ unsigned ldsoff(const void* p) {
  return (unsigned)(unsigned long long)p;   // low 32 bits of flat LDS addr = LDS offset
}

#define EDIM  1024
#define TDIM  2048
#define HNUM  16
#define HDV   64
#define FFDIM 4096

// ---------- fp32 -> bf16 convert ----------
__global__ __launch_bounds__(256) void cvt_bf16(const float* __restrict__ in,
                                                unsigned short* __restrict__ out) {
  size_t i = (size_t)blockIdx.x * 256 + threadIdx.x;   // one f32x4 per thread
  f32x4 f = ((const f32x4*)in)[i];
  u16x4 o;
  o.x = f2bf(f.x); o.y = f2bf(f.y); o.z = f2bf(f.z); o.w = f2bf(f.w);
  ((u16x4*)out)[i] = o;
}

// ---------- LayerNorm (row of 1024) -> bf16 ----------
__global__ __launch_bounds__(256) void ln_bf16(const float* __restrict__ x,
                                               const float* __restrict__ g,
                                               const float* __restrict__ be,
                                               unsigned short* __restrict__ out) {
  __shared__ float red[16];
  int row = blockIdx.x, tid = threadIdx.x;
  f32x4 v = ((const f32x4*)(x + (size_t)row * EDIM))[tid];
  float s  = v.x + v.y + v.z + v.w;
  float s2 = v.x*v.x + v.y*v.y + v.z*v.z + v.w*v.w;
#pragma unroll
  for (int d = 16; d; d >>= 1) { s += __shfl_xor(s, d, 32); s2 += __shfl_xor(s2, d, 32); }
  int wave = tid >> 5, lane = tid & 31;
  if (lane == 0) { red[wave] = s; red[8 + wave] = s2; }
  __syncthreads();
  float ts = 0.f, ts2 = 0.f;
#pragma unroll
  for (int w = 0; w < 8; ++w) { ts += red[w]; ts2 += red[8 + w]; }
  float mean = ts * (1.f / EDIM);
  float var  = ts2 * (1.f / EDIM) - mean * mean;
  float rstd = rsqrtf(var + 1e-5f);
  int c = tid * 4;
  u16x4 o;
  o.x = f2bf((v.x - mean) * rstd * g[c + 0] + be[c + 0]);
  o.y = f2bf((v.y - mean) * rstd * g[c + 1] + be[c + 1]);
  o.z = f2bf((v.z - mean) * rstd * g[c + 2] + be[c + 2]);
  o.w = f2bf((v.w - mean) * rstd * g[c + 3] + be[c + 3]);
  ((u16x4*)(out + (size_t)row * EDIM))[tid] = o;
}

// ---------- bf16 WMMA GEMM: C[M,N] = A[M,K] * Bt[N,K]^T (+bias,+relu,+resid) ----------
// flags: 1 = relu, 2 = fp32 output, 4 = add fp32 residual (requires fp32 out)
// 128x256 block tile, 64x64 wave tile (16 WMMA / wave / k-step),
// double-buffered LDS fed by async global->LDS copies; last k-tile peeled so the
// steady-state loop is branch-free: stage -> s_wait_asynccnt 6 -> barrier -> WMMAs.
#define BM 128
#define BN 256
#define BK 32
#define LDT 40   // padded LDS row stride (halves); 80B keeps 16B alignment

__global__ __launch_bounds__(256) void gemm_bf16(const unsigned short* __restrict__ A,
                                                 const unsigned short* __restrict__ Bt,
                                                 const float* __restrict__ bias,
                                                 const float* __restrict__ resid,
                                                 void* __restrict__ out,
                                                 int M, int N, int K, int flags) {
  __shared__ unsigned short sA[2][BM * LDT];
  __shared__ unsigned short sB[2][BN * LDT];
  int tid = threadIdx.x;
  int wave = tid >> 5, lane = tid & 31;
  int ln = lane & 15, lh = lane >> 4;
  int wm = wave >> 2, wn = wave & 3;            // 2x4 wave grid -> 64x64 per wave
  int bm = blockIdx.y * BM, bn = blockIdx.x * BN;

  int ldr = tid >> 2;                           // staging row (A: +0/+64; B: 4 slices)
  int lds_seg = (tid & 3) << 3;                 // staging col segment (8 halves)

  v8f acc[4][4];
#pragma unroll
  for (int mt = 0; mt < 4; ++mt)
#pragma unroll
    for (int nt = 0; nt < 4; ++nt) acc[mt][nt] = vzero8();

  auto stage = [&](int k0, int bf) {
#pragma unroll
    for (int i = 0; i < 2; ++i) {
      int r = ldr + i * 64;
      async_ld_b128(ldsoff(&sA[bf][r * LDT + lds_seg]),
                    &A[(size_t)(bm + r) * K + k0 + lds_seg]);
    }
#pragma unroll
    for (int i = 0; i < 4; ++i) {
      int r = ldr + i * 64;
      async_ld_b128(ldsoff(&sB[bf][r * LDT + lds_seg]),
                    &Bt[(size_t)(bn + r) * K + k0 + lds_seg]);
    }
  };

  auto compute = [&](int cur) {
    Frag32 a[4], b[4];
#pragma unroll
    for (int mt = 0; mt < 4; ++mt) {
      const unsigned short* p = &sA[cur][(wm * 64 + mt * 16 + ln) * LDT + lh * 8];
      a[mt].u[0] = *(const u32x4*)p;
      a[mt].u[1] = *(const u32x4*)(p + 16);
    }
#pragma unroll
    for (int nt = 0; nt < 4; ++nt) {
      const unsigned short* p = &sB[cur][(wn * 64 + nt * 16 + ln) * LDT + lh * 8];
      b[nt].u[0] = *(const u32x4*)p;
      b[nt].u[1] = *(const u32x4*)(p + 16);
    }
#pragma unroll
    for (int nt = 0; nt < 4; ++nt)
#pragma unroll
      for (int mt = 0; mt < 4; ++mt)
        acc[mt][nt] = __builtin_amdgcn_wmma_f32_16x16x32_bf16(
            false, a[mt].v, false, b[nt].v, (short)0, acc[mt][nt], false, false);
  };

  int ktiles = K / BK;
  stage(0, 0);
  for (int kt = 0; kt < ktiles - 1; ++kt) {     // steady state: branch-free body
    int cur = kt & 1;
    stage((kt + 1) * BK, cur ^ 1);
    WAIT_ASYNC(6);                              // in-order: tile kt's 6 ops done
    __syncthreads();
    compute(cur);
    __syncthreads();                            // protect buffer before next overwrite
  }
  WAIT_ASYNC(0);                                // last tile
  __syncthreads();
  compute((ktiles - 1) & 1);

#pragma unroll
  for (int mt = 0; mt < 4; ++mt)
#pragma unroll
    for (int nt = 0; nt < 4; ++nt) {
      int col = bn + wn * 64 + nt * 16 + ln;
      float bv = bias[col];
#pragma unroll
      for (int r = 0; r < 8; ++r) {
        int row = bm + wm * 64 + mt * 16 + lh * 8 + r;
        float val = acc[mt][nt][r] + bv;
        if (flags & 1) val = fmaxf(val, 0.f);
        size_t o = (size_t)row * N + col;
        if (flags & 2) {
          if (flags & 4) val += resid[o];
          ((float*)out)[o] = val;
        } else {
          ((unsigned short*)out)[o] = f2bf(val);
        }
      }
    }
}

// ---------- causal flash attention, bf16 WMMA ----------
// grid: x = b*H + h (32), y = q-tile of 64 rows (32). 128 threads, wave w owns 16 q rows.
__global__ __launch_bounds__(128) void attn_fa(const unsigned short* __restrict__ Q,
                                               const unsigned short* __restrict__ K,
                                               const unsigned short* __restrict__ V,
                                               unsigned short* __restrict__ O) {
  __shared__ unsigned short sVt[64 * LDT];          // V^T tile: [hd=64][kv=32 (+pad)]
  __shared__ unsigned short sP[4 * 16 * LDT];       // per-wave P tile [16][32 (+pad)]
  int tid = threadIdx.x;
  int wave = tid >> 5, lane = tid & 31;
  int ln = lane & 15, lh = lane >> 4;
  int b = blockIdx.x >> 4, h = blockIdx.x & 15;
  int qt = blockIdx.y;
  int g0 = qt * 64 + wave * 16;                     // wave's first q row (within T)
  size_t rowbase = (size_t)b * TDIM + g0;

  // Q fragments (HD=64 -> two K=32 steps)
  Frag32 aQ[2];
#pragma unroll
  for (int ks = 0; ks < 2; ++ks) {
    const unsigned short* p = &Q[(rowbase + ln) * EDIM + h * HDV + ks * 32 + lh * 8];
    aQ[ks].u[0] = *(const u32x4*)p;
    aQ[ks].u[1] = *(const u32x4*)(p + 16);
  }

  float mrow[8], lrow[8];
  v8f oacc[4];
#pragma unroll
  for (int r = 0; r < 8; ++r) { mrow[r] = -3.0e38f; lrow[r] = 0.f; }
#pragma unroll
  for (int nt = 0; nt < 4; ++nt) oacc[nt] = vzero8();

  int jcount = qt * 2 + 2;                          // kv tiles of 32 up to block diagonal
  unsigned short* pw = &sP[wave * 16 * LDT];

  for (int j = 0; j < jcount; ++j) {
    // cooperative stage of V^T: thread -> (kv row t', 16 hd cols)
    {
      int tp = tid >> 2;                            // kv row 0..31
      int cs = (tid & 3) * 16;                      // hd col segment
      Frag32 f;
      const u32x4* vp = (const u32x4*)&V[((size_t)b * TDIM + j * 32 + tp) * EDIM + h * HDV + cs];
      f.u[0] = vp[0];
      f.u[1] = vp[1];
#pragma unroll
      for (int i = 0; i < 16; ++i) sVt[(cs + i) * LDT + tp] = f.s[i];
    }
    __syncthreads();

    if (j * 32 <= g0 + 15) {                        // wave has unmasked work in this tile
      // S = Q * K^T  (16 x 32), two N-tiles, two K-steps each
      v8f s[2];
      s[0] = vzero8(); s[1] = vzero8();
#pragma unroll
      for (int nt = 0; nt < 2; ++nt)
#pragma unroll
        for (int ks = 0; ks < 2; ++ks) {
          Frag32 bk;
          const unsigned short* p =
              &K[((size_t)b * TDIM + j * 32 + nt * 16 + ln) * EDIM + h * HDV + ks * 32 + lh * 8];
          bk.u[0] = *(const u32x4*)p;
          bk.u[1] = *(const u32x4*)(p + 16);
          s[nt] = __builtin_amdgcn_wmma_f32_16x16x32_bf16(
              false, aQ[ks].v, false, bk.v, (short)0, s[nt], false, false);
        }
      // scale + causal mask
#pragma unroll
      for (int nt = 0; nt < 2; ++nt)
#pragma unroll
        for (int r = 0; r < 8; ++r) {
          int rowg = g0 + lh * 8 + r;
          int colg = j * 32 + nt * 16 + ln;
          float val = s[nt][r] * 0.125f;            // 1/sqrt(64)
          if (colg > rowg) val = -3.0e38f;
          s[nt][r] = val;
        }
      // online softmax per row (rows live in 16-lane halves; xor 1/2/4/8 stays in-half)
#pragma unroll
      for (int r = 0; r < 8; ++r) {
        float mx = fmaxf(s[0][r], s[1][r]);
#pragma unroll
        for (int d = 1; d < 16; d <<= 1) mx = fmaxf(mx, __shfl_xor(mx, d, 32));
        float mnew = fmaxf(mrow[r], mx);
        float corr = __expf(mrow[r] - mnew);
        float p0 = __expf(s[0][r] - mnew);
        float p1 = __expf(s[1][r] - mnew);
        float rs = p0 + p1;
#pragma unroll
        for (int d = 1; d < 16; d <<= 1) rs += __shfl_xor(rs, d, 32);
        lrow[r] = lrow[r] * corr + rs;
        mrow[r] = mnew;
#pragma unroll
        for (int nt = 0; nt < 4; ++nt) oacc[nt][r] *= corr;
        s[0][r] = p0; s[1][r] = p1;
      }
      // P (C-layout) -> LDS -> A-fragment layout
#pragma unroll
      for (int nt = 0; nt < 2; ++nt)
#pragma unroll
        for (int r = 0; r < 8; ++r)
          pw[(lh * 8 + r) * LDT + nt * 16 + ln] = f2bf(s[nt][r]);
      Frag32 aP;
      {
        const unsigned short* p = &pw[ln * LDT + lh * 8];
        aP.u[0] = *(const u32x4*)p;
        aP.u[1] = *(const u32x4*)(p + 16);
      }
      // O += P * V  (V^T staged in LDS as [hd][kv] = N x K)
#pragma unroll
      for (int nt = 0; nt < 4; ++nt) {
        Frag32 bv;
        const unsigned short* p = &sVt[(nt * 16 + ln) * LDT + lh * 8];
        bv.u[0] = *(const u32x4*)p;
        bv.u[1] = *(const u32x4*)(p + 16);
        oacc[nt] = __builtin_amdgcn_wmma_f32_16x16x32_bf16(
            false, aP.v, false, bv.v, (short)0, oacc[nt], false, false);
      }
    }
    __syncthreads();
  }

  // normalize + store bf16
#pragma unroll
  for (int r = 0; r < 8; ++r) {
    float inv = 1.f / lrow[r];
    size_t row = rowbase + lh * 8 + r;
#pragma unroll
    for (int nt = 0; nt < 4; ++nt)
      O[row * EDIM + h * HDV + nt * 16 + ln] = f2bf(oacc[nt][r] * inv);
  }
}

// ---------- host launch ----------
extern "C" void kernel_launch(void* const* d_in, const int* in_sizes, int n_in,
                              void* d_out, int out_size, void* d_ws, size_t ws_size,
                              hipStream_t stream) {
  const float* x    = (const float*)d_in[0];
  const float* Wq   = (const float*)d_in[1];
  const float* bq   = (const float*)d_in[2];
  const float* Wk   = (const float*)d_in[3];
  const float* bk   = (const float*)d_in[4];
  const float* Wv   = (const float*)d_in[5];
  const float* bv   = (const float*)d_in[6];
  const float* Wo   = (const float*)d_in[7];
  const float* bo   = (const float*)d_in[8];
  const float* W1   = (const float*)d_in[9];
  const float* b1   = (const float*)d_in[10];
  const float* W2   = (const float*)d_in[11];
  const float* b2   = (const float*)d_in[12];
  const float* ln1g = (const float*)d_in[13];
  const float* ln1b = (const float*)d_in[14];
  const float* ln2g = (const float*)d_in[15];
  const float* ln2b = (const float*)d_in[16];

  const size_t MB = 1ull << 20;
  char* ws = (char*)d_ws;
  unsigned short* wqb = (unsigned short*)(ws + 0 * MB);    // 2 MB
  unsigned short* wkb = (unsigned short*)(ws + 2 * MB);    // 2 MB
  unsigned short* wvb = (unsigned short*)(ws + 4 * MB);    // 2 MB
  unsigned short* wob = (unsigned short*)(ws + 6 * MB);    // 2 MB
  unsigned short* w1b = (unsigned short*)(ws + 8 * MB);    // 8 MB
  unsigned short* w2b = (unsigned short*)(ws + 16 * MB);   // 8 MB
  unsigned short* hb  = (unsigned short*)(ws + 24 * MB);   // 8 MB
  unsigned short* qb  = (unsigned short*)(ws + 32 * MB);   // 8 MB
  unsigned short* kb  = (unsigned short*)(ws + 40 * MB);   // 8 MB
  unsigned short* vb  = (unsigned short*)(ws + 48 * MB);   // 8 MB
  unsigned short* ab  = (unsigned short*)(ws + 56 * MB);   // 8 MB (attention out)
  float*          x2  = (float*)(ws + 64 * MB);            // 16 MB
  unsigned short* h2b = (unsigned short*)(ws + 80 * MB);   // 8 MB
  unsigned short* f1  = (unsigned short*)(ws + 24 * MB);   // 32 MB, reuses hb/q/k/v

  const int M = 2 * TDIM;                                   // 4096 rows

  // weights -> bf16
  cvt_bf16<<<(EDIM * EDIM) / 1024, 256, 0, stream>>>(Wq, wqb);
  cvt_bf16<<<(EDIM * EDIM) / 1024, 256, 0, stream>>>(Wk, wkb);
  cvt_bf16<<<(EDIM * EDIM) / 1024, 256, 0, stream>>>(Wv, wvb);
  cvt_bf16<<<(EDIM * EDIM) / 1024, 256, 0, stream>>>(Wo, wob);
  cvt_bf16<<<(FFDIM * EDIM) / 1024, 256, 0, stream>>>(W1, w1b);
  cvt_bf16<<<(EDIM * FFDIM) / 1024, 256, 0, stream>>>(W2, w2b);

  // LN1
  ln_bf16<<<M, 256, 0, stream>>>(x, ln1g, ln1b, hb);

  // Q, K, V projections
  dim3 gE(EDIM / BN, M / BM);
  gemm_bf16<<<gE, 256, 0, stream>>>(hb, wqb, bq, nullptr, qb, M, EDIM, EDIM, 0);
  gemm_bf16<<<gE, 256, 0, stream>>>(hb, wkb, bk, nullptr, kb, M, EDIM, EDIM, 0);
  gemm_bf16<<<gE, 256, 0, stream>>>(hb, wvb, bv, nullptr, vb, M, EDIM, EDIM, 0);

  // attention
  attn_fa<<<dim3(2 * HNUM, TDIM / 64), 128, 0, stream>>>(qb, kb, vb, ab);

  // O projection + residual -> x2 (fp32)
  gemm_bf16<<<gE, 256, 0, stream>>>(ab, wob, bo, x, x2, M, EDIM, EDIM, 2 | 4);

  // LN2
  ln_bf16<<<M, 256, 0, stream>>>(x2, ln2g, ln2b, h2b);

  // FFN
  dim3 gF1(FFDIM / BN, M / BM);
  gemm_bf16<<<gF1, 256, 0, stream>>>(h2b, w1b, b1, nullptr, f1, M, FFDIM, EDIM, 1);
  gemm_bf16<<<gE, 256, 0, stream>>>(f1, w2b, b2, x2, (float*)d_out, M, EDIM, FFDIM, 2 | 4);
}
